// get_qkv_53575422050708
// MI455X (gfx1250) — matrix operand order repaired and checked by reading, not compiled, not run on hardware
//
#include <hip/hip_runtime.h>
#include <hip/hip_bf16.h>
#include <stdint.h>

typedef __bf16 bf16_t;
typedef __attribute__((ext_vector_type(16))) __bf16 v16bf;
typedef __attribute__((ext_vector_type(8)))  float  v8f;

#define DIMK  1024
#define NQKV  3072
#define MROWS 32768
#define SEQ   4096
#define HEADS 16
#define DHEAD 64

union FragU { v16bf v; uint4 q[2]; };

__device__ __forceinline__ unsigned short f2bf_rne(float f) {
  union { float f; uint32_t u; } c; c.f = f;
  uint32_t u = c.u;
  uint32_t r = u + 0x7FFFu + ((u >> 16) & 1u);   // round-to-nearest-even
  return (unsigned short)(r >> 16);
}

// fp32 -> bf16 conversion pre-pass (pure bandwidth; halves X traffic for the
// GEMM and shrinks W to 6MB so it pins in the 192MB L2).
__global__ __launch_bounds__(256) void cvt_f32_bf16(const float* __restrict__ in,
                                                    unsigned short* __restrict__ out,
                                                    int n4) {
  int i = blockIdx.x * blockDim.x + threadIdx.x;
  if (i >= n4) return;
  float4 f = reinterpret_cast<const float4*>(in)[i];
  ushort4 o;
  o.x = f2bf_rne(f.x); o.y = f2bf_rne(f.y);
  o.z = f2bf_rne(f.z); o.w = f2bf_rne(f.w);
  reinterpret_cast<ushort4*>(out)[i] = o;
}

// C[m, e] = sum_k X[m,k] * W[e,k]; output scattered to [3][b][h][n][d].
// Block: 256 threads = 8 wave32 arranged 4(M) x 2(N); block tile 128x128.
// Wave tile 32x64 = 2x4 accumulators of 16x16x32 bf16 WMMA:
//   A frags reused 4x, B frags 2x -> 21.3 FLOP per fragment byte.
__global__ __launch_bounds__(256) void qkv_wmma_bf16(const bf16_t* __restrict__ X,
                                                     const bf16_t* __restrict__ Wb,
                                                     float* __restrict__ out) {
  const int lane   = threadIdx.x & 31;
  const int wave   = threadIdx.x >> 5;
  const int wave_m = wave & 3;           // 4 M-waves * 32 rows = 128
  const int wave_n = wave >> 2;          // 2 N-waves * 64 cols = 128
  const int l16    = lane & 15;
  const int hlf    = lane >> 4;          // 0: lanes 0-15, 1: lanes 16-31

  const int m0 = blockIdx.y * 128 + wave_m * 32;
  const int n0 = blockIdx.x * 128 + wave_n * 64;

  // A fragment (16-bit A 16x32): lane row = m0+l16; low half-wave holds
  // K 0..15 contiguous, high half-wave K 16..31.
  const bf16_t* Abase = X  + (size_t)(m0 + l16) * DIMK + hlf * 16;
  // B fragment (16-bit B 32x16): lane col = n0+l16; low half-wave holds
  // K 0..15 contiguous, high half-wave K 16..31.
  const bf16_t* Bbase = Wb + (size_t)(n0 + l16) * DIMK + hlf * 16;

  v8f acc[2][4];
#pragma unroll
  for (int i = 0; i < 2; ++i)
#pragma unroll
    for (int j = 0; j < 4; ++j) acc[i][j] = (v8f){};

#pragma unroll 2
  for (int kk = 0; kk < DIMK; kk += 32) {
    // Prefetch the next K-chunk of each fragment row (global_prefetch_b8);
    // speculative prefetches past the tensor edge are dropped by HW.
    __builtin_prefetch(Abase + kk + 64, 0, 3);
    __builtin_prefetch(Bbase + kk + 64, 0, 3);

    FragU a[2], b[4];
#pragma unroll
    for (int i = 0; i < 2; ++i) {
      const bf16_t* p = Abase + (size_t)i * 16 * DIMK + kk;
      a[i].q[0] = *reinterpret_cast<const uint4*>(p);
      a[i].q[1] = *reinterpret_cast<const uint4*>(p + 8);
    }
#pragma unroll
    for (int j = 0; j < 4; ++j) {
      const bf16_t* p = Bbase + (size_t)j * 16 * DIMK + kk;
      b[j].q[0] = *reinterpret_cast<const uint4*>(p);
      b[j].q[1] = *reinterpret_cast<const uint4*>(p + 8);
    }
#pragma unroll
    for (int i = 0; i < 2; ++i)
#pragma unroll
      for (int j = 0; j < 4; ++j)
        acc[i][j] = __builtin_amdgcn_wmma_f32_16x16x32_bf16(
            false, a[i].v, false, b[j].v, (short)0, acc[i][j], false, false);
  }

  // C/D layout: VGPR v, lane l -> M = v + 8*(l>>4), N = l&15.
  // e = global col: which = e>>10 (q/k/v), h = (e>>6)&15, d = e&63.
  // A 16-wide tile never crosses a head (64 % 16 == 0) -> coalesced stores.
  auto store_tile = [&](v8f accv, int mbase, int nbase) {
    int e      = nbase + l16;
    int which  = e >> 10;
    int hd     = e & 1023;
    int h      = hd >> 6;
    int d      = hd & 63;
    int mfirst = mbase + hlf * 8;
    int bb     = mfirst >> 12;             // 4096 rows per batch, tiles aligned
    size_t base = ((size_t)(which * 8 + bb) * HEADS + h) * (size_t)(SEQ * DHEAD) + d;
#pragma unroll
    for (int v = 0; v < 8; ++v) {
      int nseq = (mfirst + v) & (SEQ - 1);
      out[base + (size_t)nseq * DHEAD] = accv[v];
    }
  };
#pragma unroll
  for (int i = 0; i < 2; ++i)
#pragma unroll
    for (int j = 0; j < 4; ++j)
      store_tile(acc[i][j], m0 + i * 16, n0 + j * 16);
}

extern "C" void kernel_launch(void* const* d_in, const int* in_sizes, int n_in,
                              void* d_out, int out_size, void* d_ws, size_t ws_size,
                              hipStream_t stream) {
  const float* x = (const float*)d_in[0];   // [8, 4096, 1024]
  const float* W = (const float*)d_in[1];   // [3072, 1024]
  float* out = (float*)d_out;               // 3 x [8, 16, 4096, 64] concat

  // Workspace: bf16 copies of X (64MB) then W (6MB).
  size_t xElems = (size_t)in_sizes[0];
  size_t wElems = (size_t)in_sizes[1];
  unsigned short* Xb = (unsigned short*)d_ws;
  unsigned short* Wc = Xb + xElems;

  int nX4 = (int)(xElems / 4);
  int nW4 = (int)(wElems / 4);
  cvt_f32_bf16<<<(nX4 + 255) / 256, 256, 0, stream>>>(x, Xb, nX4);
  cvt_f32_bf16<<<(nW4 + 255) / 256, 256, 0, stream>>>(W, Wc, nW4);

  dim3 grid(NQKV / 128, MROWS / 128);
  qkv_wmma_bf16<<<grid, dim3(256), 0, stream>>>((const bf16_t*)Xb,
                                                (const bf16_t*)Wc, out);
}
